// GAT_Linear_80642305949902
// MI455X (gfx1250) — matrix operand order repaired and checked by reading, not compiled, not run on hardware
//
#include <hip/hip_runtime.h>
#include <hip/hip_bf16.h>
#include <math.h>

typedef __attribute__((ext_vector_type(16))) _Float16 v16h;
typedef __attribute__((ext_vector_type(8)))  float    v8f;

#define NEG_SLOPE 0.2f

__device__ __forceinline__ float lrelu(float v) { return v > 0.f ? v : NEG_SLOPE * v; }

// float atomic max via signed/unsigned integer atomics (valid with -inf init)
__device__ __forceinline__ void atomicMaxF32(float* addr, float v) {
  if (v >= 0.f) atomicMax((int*)addr, __float_as_int(v));
  else          atomicMin((unsigned int*)addr, __float_as_uint(v));
}

// ---------------------------------------------------------------- utilities
__global__ void k_fill(float* __restrict__ p, float v, int n) {
  int i = blockIdx.x * blockDim.x + threadIdx.x;
  if (i < n) p[i] = v;
}

__global__ void k_f32_to_f16(const float* __restrict__ in, _Float16* __restrict__ out, int n) {
  int i = blockIdx.x * blockDim.x + threadIdx.x;
  if (i < n) out[i] = (_Float16)in[i];
}

// W[K][N] (row-major) -> Wt[N][K] f16
__global__ void k_transpose_f16(const float* __restrict__ W, _Float16* __restrict__ Wt,
                                int K, int N) {
  int i = blockIdx.x * blockDim.x + threadIdx.x;
  if (i >= K * N) return;
  int k = i / N, n = i - k * N;
  Wt[(size_t)n * K + k] = (_Float16)W[i];
}

// ---------------------------------------------------------------- WMMA GEMM
// C[M,N] = A[M,K]_f16 * Bt[N,K]_f16^T   (fp32 accumulate)
// one wave -> 16 rows x (16*NT) cols; K multiple of 32; N multiple of 16*NT.
template <int NT>
__global__ void k_gemm_wmma(const _Float16* __restrict__ A,
                            const _Float16* __restrict__ Bt,
                            float* __restrict__ C,
                            int M, int N, int K) {
  union Frag { v16h h; float4 f[2]; };
  const int lane  = threadIdx.x & 31;
  const int wave  = threadIdx.x >> 5;
  const int mtile = blockIdx.x * (blockDim.x >> 5) + wave;
  const int m0    = mtile * 16;
  if (m0 >= M) return;                       // whole-wave exit: EXEC stays all-ones
  const int n0  = blockIdx.y * (16 * NT);
  const int r16 = lane & 15;
  const int kg  = lane >> 4;

  v8f acc[NT];
  for (int t = 0; t < NT; ++t)
    for (int r = 0; r < 8; ++r) acc[t][r] = 0.0f;

  const int arow = (m0 + r16 < M) ? (m0 + r16) : (M - 1);
  const _Float16* Ap = A + (size_t)arow * K;

  for (int kb = 0; kb < K; kb += 32) {
    // 16-bit A 16x32 layout: lane m holds K = kg*8 + [0..7] and 16 + kg*8 + [0..7]
    Frag a;
    a.f[0] = *reinterpret_cast<const float4*>(Ap + kb + kg * 8);
    a.f[1] = *reinterpret_cast<const float4*>(Ap + kb + 16 + kg * 8);
#pragma unroll
    for (int t = 0; t < NT; ++t) {
      // 16-bit B 32x16 layout: lane holds col n, K = kg*8 + [0..7] and 16 + kg*8 + [0..7]
      const _Float16* Bp = Bt + (size_t)(n0 + t * 16 + r16) * K + kb + kg * 8;
      Frag b;
      b.f[0] = *reinterpret_cast<const float4*>(Bp);
      b.f[1] = *reinterpret_cast<const float4*>(Bp + 16);
      acc[t] = __builtin_amdgcn_wmma_f32_16x16x32_f16(
          false, a.h, false, b.h, (short)0, acc[t], false, false);
    }
  }

#pragma unroll
  for (int t = 0; t < NT; ++t)
#pragma unroll
    for (int r = 0; r < 8; ++r) {
      int row = m0 + r + 8 * kg;             // C/D layout: VGPR r -> M = r + 8*kg
      if (row < M) C[(size_t)row * N + n0 + t * 16 + r16] = acc[t][r];
    }
}

// ---------------------------------------------------------------- attention
// a_src[n,h] = sum_c Hf[n, h*C+c]*att_s[h,c] ; same for a_dst
__global__ void k_attn_coeff(const float* __restrict__ Hf,
                             const float* __restrict__ att_s,
                             const float* __restrict__ att_d,
                             float* __restrict__ a_s, float* __restrict__ a_d,
                             int Nn, int H, int C) {
  int i = blockIdx.x * blockDim.x + threadIdx.x;
  if (i >= Nn * H) return;
  int n = i / H, h = i - n * H;
  const float* row = Hf + (size_t)n * (H * C) + h * C;
  float ss = 0.f, sd = 0.f;
  for (int c = 0; c < C; ++c) {
    float v = row[c];
    ss += v * att_s[h * C + c];
    sd += v * att_d[h * C + c];
  }
  a_s[i] = ss;
  a_d[i] = sd;
}

// ---------------------------------------------------------------- edge passes
// ei: [src(E), dst(E)]; edges e>=E are self loops (s=d=e-E)
__global__ void k_edge_max(const int* __restrict__ ei, int E, int Nn, int H,
                           const float* __restrict__ as_, const float* __restrict__ ad_,
                           float* __restrict__ emax) {
  int i = blockIdx.x * blockDim.x + threadIdx.x;
  int tot = (E + Nn) * H;
  if (i >= tot) return;
  int e = i / H, h = i - e * H;
  int s, d;
  if (e < E) { s = ei[e]; d = ei[E + e]; } else { s = d = e - E; }
  float v = lrelu(as_[s * H + h] + ad_[d * H + h]);
  atomicMaxF32(&emax[d * H + h], v);
}

__global__ void k_edge_expsum(const int* __restrict__ ei, int E, int Nn, int H,
                              const float* __restrict__ as_, const float* __restrict__ ad_,
                              const float* __restrict__ emax, float* __restrict__ den) {
  int i = blockIdx.x * blockDim.x + threadIdx.x;
  int tot = (E + Nn) * H;
  if (i >= tot) return;
  int e = i / H, h = i - e * H;
  int s, d;
  if (e < E) { s = ei[e]; d = ei[E + e]; } else { s = d = e - E; }
  int dh = d * H + h;
  float v = lrelu(as_[s * H + h] + ad_[dh]);
  atomicAdd(&den[dh], __expf(v - emax[dh]));
}

// one thread = (edge, 4 channels): float4 gather + 4 float atomic scatters
__global__ void k_edge_agg(const int* __restrict__ ei, int E, int Nn, int H, int C,
                           const float* __restrict__ as_, const float* __restrict__ ad_,
                           const float* __restrict__ emax, const float* __restrict__ den,
                           const float* __restrict__ Hf, float* __restrict__ outp) {
  int i = blockIdx.x * blockDim.x + threadIdx.x;
  int chunks = (H * C) >> 2;
  long long tot = (long long)(E + Nn) * chunks;
  if (i >= tot) return;
  int e   = i / chunks;
  int ch4 = (i - e * chunks) << 2;
  int h   = ch4 / C;
  int s, d;
  if (e < E) { s = ei[e]; d = ei[E + e]; } else { s = d = e - E; }
  int dh = d * H + h;
  float v     = lrelu(as_[s * H + h] + ad_[dh]);
  float alpha = __expf(v - emax[dh]) / (den[dh] + 1e-16f);
  const float4 m = *reinterpret_cast<const float4*>(Hf + (size_t)s * (H * C) + ch4);
  float* op = outp + (size_t)d * (H * C) + ch4;
  atomicAdd(op + 0, m.x * alpha);
  atomicAdd(op + 1, m.y * alpha);
  atomicAdd(op + 2, m.z * alpha);
  atomicAdd(op + 3, m.w * alpha);
}

// bias + ReLU + fp32->f16 (layer-2 GEMM input)
__global__ void k_finalize_relu_f16(const float* __restrict__ agg,
                                    const float* __restrict__ bias,
                                    _Float16* __restrict__ out16, int n, int CH) {
  int i = blockIdx.x * blockDim.x + threadIdx.x;
  if (i >= n) return;
  float v = agg[i] + bias[i % CH];
  out16[i] = (_Float16)fmaxf(v, 0.f);
}

// ---------------------------------------------------------------- pool + MLP
__global__ void k_pool(const float* __restrict__ h2, const float* __restrict__ bias,
                       const int* __restrict__ batch,
                       float* __restrict__ pooled, float* __restrict__ cnt,
                       int Nn, int C) {
  int i = blockIdx.x * blockDim.x + threadIdx.x;
  if (i >= Nn * C) return;
  int n = i / C, c = i - n * C;
  int g = batch[n];
  atomicAdd(&pooled[g * C + c], h2[i] + bias[c]);
  if (c == 0) atomicAdd(&cnt[g], 1.0f);
}

__global__ void k_head(const float* __restrict__ pooled, const float* __restrict__ cnt,
                       const float* __restrict__ w1, const float* __restrict__ b1,
                       const float* __restrict__ w2, const float* __restrict__ b2,
                       float* __restrict__ out, int NG, int C, int HM, int O) {
  int g = threadIdx.x;
  if (g >= NG) return;
  float v[32], t[16];
  float cn = fmaxf(cnt[g], 1.0f);
  for (int c = 0; c < C; ++c) v[c] = fmaxf(pooled[g * C + c] / cn, 0.f);
  for (int j = 0; j < HM; ++j) {
    float s = b1[j];
    for (int c = 0; c < C; ++c) s += v[c] * w1[c * HM + j];
    t[j] = fmaxf(s, 0.f);
  }
  for (int k = 0; k < O; ++k) {
    float s = b2[k];
    for (int j = 0; j < HM; ++j) s += t[j] * w2[j * O + k];
    out[g * O + k] = fmaxf(s, 0.f);
  }
}

// ---------------------------------------------------------------- launch
static inline int cdiv(long long a, long long b) { return (int)((a + b - 1) / b); }

extern "C" void kernel_launch(void* const* d_in, const int* in_sizes, int n_in,
                              void* d_out, int out_size, void* d_ws, size_t ws_size,
                              hipStream_t stream) {
  const int NN = 50000, E = 800000, IND = 256, HIDC = 64, HEADS = 4, OUTD = 16, NG = 64;
  const int CH1 = HEADS * HIDC;  // 256
  const int C2  = HIDC / 2;      // 32
  const int HM  = HIDC / 4;      // 16
  const int B   = 256;           // 8 wave32s per block
  (void)in_sizes; (void)n_in; (void)out_size; (void)ws_size;

  const float* x   = (const float*)d_in[0];
  const int*   ei  = (const int*)d_in[1];
  const int*   bat = (const int*)d_in[2];
  const float* W1  = (const float*)d_in[3];
  const float* as1 = (const float*)d_in[4];
  const float* ad1 = (const float*)d_in[5];
  const float* b1  = (const float*)d_in[6];
  const float* W2  = (const float*)d_in[7];
  const float* as2 = (const float*)d_in[8];
  const float* ad2 = (const float*)d_in[9];
  const float* b2  = (const float*)d_in[10];
  const float* f1w = (const float*)d_in[11];
  const float* f1b = (const float*)d_in[12];
  const float* f2w = (const float*)d_in[13];
  const float* f2b = (const float*)d_in[14];
  float* out = (float*)d_out;

  // workspace layout (256B aligned), with reuse
  char* ws = (char*)d_ws;
  size_t off = 0;
  auto take = [&](size_t bytes) -> char* {
    char* p = ws + off;
    off = (off + bytes + 255) & ~(size_t)255;
    return p;
  };
  _Float16* Xh  = (_Float16*)take((size_t)NN * CH1 * 2);  // reused as layer-2 f16 input
  _Float16* W1t = (_Float16*)take((size_t)CH1 * IND * 2);
  _Float16* W2t = (_Float16*)take((size_t)C2 * CH1 * 2);
  float* H1   = (float*)take((size_t)NN * CH1 * 4);       // reused as G2 [NN x 32]
  float* OUT1 = (float*)take((size_t)NN * CH1 * 4);       // reused as OUT2 [NN x 32]
  float* ASRC = (float*)take((size_t)NN * HEADS * 4);
  float* ADST = (float*)take((size_t)NN * HEADS * 4);
  float* EMAX = (float*)take((size_t)NN * HEADS * 4);
  float* DEN  = (float*)take((size_t)NN * HEADS * 4);
  float* POOL = (float*)take((size_t)NG * C2 * 4);
  float* CNT  = (float*)take((size_t)NG * 4);

  const float NEG_INF = -3.402823466e38f;
  const int ET = E + NN;  // edges incl. self loops

  // ---- prep: f16 conversions
  k_f32_to_f16<<<cdiv((long long)NN * IND, B), B, 0, stream>>>(x, Xh, NN * IND);
  k_transpose_f16<<<cdiv((long long)IND * CH1, B), B, 0, stream>>>(W1, W1t, IND, CH1);
  k_transpose_f16<<<cdiv((long long)CH1 * C2, B), B, 0, stream>>>(W2, W2t, CH1, C2);

  // ---- layer 1 GEMM: H1 = Xh @ W1   (M=50000, N=256, K=256)
  {
    dim3 grid(cdiv(NN / 16, 8), CH1 / 64);
    k_gemm_wmma<4><<<grid, B, 0, stream>>>(Xh, W1t, H1, NN, CH1, IND);
  }
  k_attn_coeff<<<cdiv((long long)NN * HEADS, B), B, 0, stream>>>(H1, as1, ad1, ASRC, ADST, NN, HEADS, HIDC);

  // ---- layer 1 segment softmax + aggregate
  k_fill<<<cdiv((long long)NN * HEADS, B), B, 0, stream>>>(EMAX, NEG_INF, NN * HEADS);
  k_fill<<<cdiv((long long)NN * HEADS, B), B, 0, stream>>>(DEN, 0.f, NN * HEADS);
  k_fill<<<cdiv((long long)NN * CH1, B), B, 0, stream>>>(OUT1, 0.f, NN * CH1);
  k_edge_max<<<cdiv((long long)ET * HEADS, B), B, 0, stream>>>(ei, E, NN, HEADS, ASRC, ADST, EMAX);
  k_edge_expsum<<<cdiv((long long)ET * HEADS, B), B, 0, stream>>>(ei, E, NN, HEADS, ASRC, ADST, EMAX, DEN);
  k_edge_agg<<<cdiv((long long)ET * (CH1 / 4), B), B, 0, stream>>>(
      ei, E, NN, HEADS, HIDC, ASRC, ADST, EMAX, DEN, H1, OUT1);
  // bias + ReLU -> f16 layer-2 input (reuse Xh region)
  k_finalize_relu_f16<<<cdiv((long long)NN * CH1, B), B, 0, stream>>>(OUT1, b1, Xh, NN * CH1, CH1);

  // ---- layer 2 GEMM: G2 = relu(h1) @ W2   (M=50000, N=32, K=256), G2 reuses H1
  float* G2   = H1;
  float* OUT2 = OUT1;
  {
    dim3 grid(cdiv(NN / 16, 8), 1);
    k_gemm_wmma<2><<<grid, B, 0, stream>>>(Xh, W2t, G2, NN, C2, CH1);
  }
  k_attn_coeff<<<cdiv((long long)NN, B), B, 0, stream>>>(G2, as2, ad2, ASRC, ADST, NN, 1, C2);

  // ---- layer 2 segment softmax + aggregate (H=1, C=32)
  k_fill<<<cdiv((long long)NN, B), B, 0, stream>>>(EMAX, NEG_INF, NN);
  k_fill<<<cdiv((long long)NN, B), B, 0, stream>>>(DEN, 0.f, NN);
  k_fill<<<cdiv((long long)NN * C2, B), B, 0, stream>>>(OUT2, 0.f, NN * C2);
  k_edge_max<<<cdiv((long long)ET, B), B, 0, stream>>>(ei, E, NN, 1, ASRC, ADST, EMAX);
  k_edge_expsum<<<cdiv((long long)ET, B), B, 0, stream>>>(ei, E, NN, 1, ASRC, ADST, EMAX, DEN);
  k_edge_agg<<<cdiv((long long)ET * (C2 / 4), B), B, 0, stream>>>(
      ei, E, NN, 1, C2, ASRC, ADST, EMAX, DEN, G2, OUT2);

  // ---- global mean pool + MLP head
  k_fill<<<cdiv((long long)NG * C2, B), B, 0, stream>>>(POOL, 0.f, NG * C2);
  k_fill<<<1, 64, 0, stream>>>(CNT, 0.f, NG);
  k_pool<<<cdiv((long long)NN * C2, B), B, 0, stream>>>(OUT2, b2, bat, POOL, CNT, NN, C2);
  k_head<<<1, 64, 0, stream>>>(POOL, CNT, f1w, f1b, f2w, f2b, out, NG, C2, HM, OUTD);
}
